// Discriminator_16492674417366
// MI455X (gfx1250) — matrix operand order repaired and checked by reading, not compiled, not run on hardware
//
#include <hip/hip_runtime.h>

typedef __attribute__((ext_vector_type(16))) _Float16 v16h;
typedef __attribute__((ext_vector_type(8)))  _Float16 v8h;
typedef __attribute__((ext_vector_type(8)))  float    v8f;
typedef __attribute__((ext_vector_type(4)))  float    v4f;

// ---------- fast nonlinearities (v_exp_f32 + v_rcp_f32) ----------
__device__ __forceinline__ float fsig(float x) {
  return __builtin_amdgcn_rcpf(1.0f + __expf(-x));
}
__device__ __forceinline__ float ftanh(float x) {
  return 1.0f - 2.0f * __builtin_amdgcn_rcpf(__expf(2.0f * x) + 1.0f);
}
// zero-state LSTM cell output for one gate triple (f-gate dead: c_prev==0)
__device__ __forceinline__ float lstm_h(float gi, float gc, float go) {
  float c = fsig(gi) * ftanh(gc);
  return fsig(go) * ftanh(c);
}

__device__ __forceinline__ v8f wmma16(v16h a, v16h b, v8f c) {
  return __builtin_amdgcn_wmma_f32_16x16x32_f16(false, a, false, b, (short)0, c,
                                                false, false);
}

__device__ __forceinline__ v8f cfill(float b) {
  v8f c;
#pragma unroll
  for (int i = 0; i < 8; ++i) c[i] = b;
  return c;
}

__device__ __forceinline__ v8f combine3(v8f di, v8f dg, v8f dq) {
  v8f h;
#pragma unroll
  for (int v = 0; v < 8; ++v) h[v] = lstm_h(di[v], dg[v], dq[v]);
  return h;
}

// Build one WMMA B tile (32x16 f16) from weight W[G x KL] (row-major, torch W_ih):
// B[k][n] = (k < KL) ? W[(16*t + n)*KL + k] : 0
// lane layout: n = lane&15; half j holds k = (j&7) + 8*(lane>>4) + 16*(j>>3)
template <int KL>
__device__ __forceinline__ v16h loadB(const float* __restrict__ W, int t, int n,
                                      int g) {
  v16h b;
  const int row = (t * 16 + n) * KL;
#pragma unroll
  for (int j = 0; j < 16; ++j) {
    const int k = (j & 7) + 8 * g + 16 * (j >> 3);
    float w = (k < KL) ? W[row + k] : 0.0f;
    b[j] = (_Float16)w;
  }
  return b;
}

// tile ids: 0..2 = L1 {i,g,o}; 3..4 = L2 {i|f, g|o}; 5..7 = L3 {i,g,o};
//           8..13 = L4 {i0,i1,g0,g1,o0,o1}
__device__ __forceinline__ v16h buildB(int which, int lane,
                                       const float* __restrict__ wih1,
                                       const float* __restrict__ wih2,
                                       const float* __restrict__ wih3,
                                       const float* __restrict__ wih4) {
  const int n = lane & 15, g = lane >> 4;
  switch (which) {
    case 0:  return loadB<32>(wih1, 0, n, g);
    case 1:  return loadB<32>(wih1, 2, n, g);
    case 2:  return loadB<32>(wih1, 3, n, g);
    case 3:  return loadB<16>(wih2, 0, n, g);
    case 4:  return loadB<16>(wih2, 1, n, g);
    case 5:  return loadB<8>(wih3, 0, n, g);
    case 6:  return loadB<8>(wih3, 2, n, g);
    case 7:  return loadB<8>(wih3, 3, n, g);
    case 8:  return loadB<16>(wih4, 0, n, g);
    case 9:  return loadB<16>(wih4, 1, n, g);
    case 10: return loadB<16>(wih4, 4, n, g);
    case 11: return loadB<16>(wih4, 5, n, g);
    case 12: return loadB<16>(wih4, 6, n, g);
    default: return loadB<16>(wih4, 7, n, g);
  }
}

// D-layout tile -> row-major f16 LDS buffer of width Wd; lane holds col, rows v+8g
__device__ __forceinline__ void storeH(_Float16* hb, int Wd, int col, int g,
                                       v8f h) {
#pragma unroll
  for (int v = 0; v < 8; ++v) hb[(v + 8 * g) * Wd + col] = (_Float16)h[v];
}

// A-layout loads from row-major f16 LDS buffer (K padded with zeros)
__device__ __forceinline__ v16h loadA16(const _Float16* hb, int m, int g) {
  const v8h lo = *(const v8h*)(hb + m * 16 + 8 * g);
  v16h a = {};
#pragma unroll
  for (int j = 0; j < 8; ++j) a[j] = lo[j];
  return a;
}
__device__ __forceinline__ v16h loadA8(const _Float16* hb, int m, int g) {
  v16h a = {};
  if (g == 0) {
    const v8h lo = *(const v8h*)(hb + m * 8);
#pragma unroll
    for (int j = 0; j < 8; ++j) a[j] = lo[j];
  }
  return a;
}

__device__ __forceinline__ void lds_fence() {
  asm volatile("s_wait_dscnt 0" ::: "memory");
}

#define TPB 256

__global__ __launch_bounds__(TPB, 1) void lstm_stack_kernel(
    const float* __restrict__ x,
    const float* __restrict__ wih0, const float* __restrict__ bih0, const float* __restrict__ bhh0,
    const float* __restrict__ wih1, const float* __restrict__ bih1, const float* __restrict__ bhh1,
    const float* __restrict__ wih2, const float* __restrict__ bih2, const float* __restrict__ bhh2,
    const float* __restrict__ wih3, const float* __restrict__ bih3, const float* __restrict__ bhh3,
    const float* __restrict__ wih4, const float* __restrict__ bih4, const float* __restrict__ bhh4,
    const float* __restrict__ wout, const float* __restrict__ bout,
    float* __restrict__ out, int nTiles) {
  __shared__ __align__(16) float sW0[192];        // w0_i,w0_g,w0_o,b0_i,b0_g,b0_o
  __shared__ v16h sB[14 * 32];                    // pre-formatted WMMA B tiles
  __shared__ __align__(16) _Float16 sH[8][512];   // per-wave activation staging

  const int tid = threadIdx.x;
  if (tid < 32) {  // layer0: gate-separated weight column + bias sums
    sW0[tid]       = wih0[tid];
    sW0[32 + tid]  = wih0[64 + tid];
    sW0[64 + tid]  = wih0[96 + tid];
    sW0[96 + tid]  = bih0[tid]      + bhh0[tid];
    sW0[128 + tid] = bih0[64 + tid] + bhh0[64 + tid];
    sW0[160 + tid] = bih0[96 + tid] + bhh0[96 + tid];
  }
  for (int idx = tid; idx < 14 * 32; idx += TPB) {
    sB[idx] = buildB(idx >> 5, idx & 31, wih1, wih2, wih3, wih4);
  }
  __syncthreads();

  const int lane = tid & 31;
  const int m = lane & 15;   // row (A/D) or column (B/D) index
  const int g = lane >> 4;   // half-wave group
  const int wave = tid >> 5;
  _Float16* hb = &sH[wave][0];

  // per-lane bias sums (lane-uniform per D tile since N = lane&15)
  const float b1i = bih1[m] + bhh1[m];
  const float b1g = bih1[32 + m] + bhh1[32 + m];
  const float b1o = bih1[48 + m] + bhh1[48 + m];
  const float b20 = bih2[m] + bhh2[m];
  const float b21 = bih2[16 + m] + bhh2[16 + m];
  const float b3i = bih3[m] + bhh3[m];
  const float b3g = bih3[32 + m] + bhh3[32 + m];
  const float b3o = bih3[48 + m] + bhh3[48 + m];
  float b4[6];
  b4[0] = bih4[m] + bhh4[m];
  b4[1] = bih4[16 + m] + bhh4[16 + m];
  b4[2] = bih4[64 + m] + bhh4[64 + m];
  b4[3] = bih4[80 + m] + bhh4[80 + m];
  b4[4] = bih4[96 + m] + bhh4[96 + m];
  b4[5] = bih4[112 + m] + bhh4[112 + m];
  const float wA = wout[m];
  const float wB = wout[16 + m];
  const float bO = bout[0];

  const v4f* sw = (const v4f*)sW0;

  const int wavesTotal = gridDim.x * (TPB / 32);
  const int waveId = blockIdx.x * (TPB / 32) + wave;

  for (int tile = waveId; tile < nTiles; tile += wavesTotal) {
    asm volatile("" ::: "memory");  // keep B-tile LDS loads inside the loop
    const int base = tile * 16;
    const float xr = x[base + m];

    // ---- layer 0 (1 -> 32): computed per-lane directly in WMMA A layout
    // half j of lane (m,g) holds K = j + 8g + (j>=8 ? 8 : 0)
    v16h a;
#pragma unroll
    for (int run = 0; run < 2; ++run) {
      const int q = run * 4 + 2 * g;  // v4f index -> float k0 = run*16 + 8g
      v4f wi0 = sw[q],      wi1 = sw[q + 1];
      v4f wg0 = sw[8 + q],  wg1 = sw[8 + q + 1];
      v4f wo0 = sw[16 + q], wo1 = sw[16 + q + 1];
      v4f bi0 = sw[24 + q], bi1 = sw[24 + q + 1];
      v4f bg0 = sw[32 + q], bg1 = sw[32 + q + 1];
      v4f bo0 = sw[40 + q], bo1 = sw[40 + q + 1];
#pragma unroll
      for (int e = 0; e < 4; ++e) {
        a[run * 8 + e] = (_Float16)lstm_h(
            xr * wi0[e] + bi0[e], xr * wg0[e] + bg0[e], xr * wo0[e] + bo0[e]);
        a[run * 8 + 4 + e] = (_Float16)lstm_h(
            xr * wi1[e] + bi1[e], xr * wg1[e] + bg1[e], xr * wo1[e] + bo1[e]);
      }
    }

    // ---- layer 1 (32 -> 16): gate tiles i, g, o
    v8f d_i = wmma16(a, sB[0 * 32 + lane], cfill(b1i));
    v8f d_g = wmma16(a, sB[1 * 32 + lane], cfill(b1g));
    v8f d_o = wmma16(a, sB[2 * 32 + lane], cfill(b1o));
    v8f h1 = combine3(d_i, d_g, d_o);
    storeH(hb, 16, m, g, h1);
    lds_fence();
    a = loadA16(hb, m, g);

    // ---- layer 2 (16 -> 8): tile0 = i|f, tile1 = g|o (o via shfl_xor 8)
    v8f d20 = wmma16(a, sB[3 * 32 + lane], cfill(b20));
    v8f d21 = wmma16(a, sB[4 * 32 + lane], cfill(b21));
    v8f h2;
#pragma unroll
    for (int v = 0; v < 8; ++v) {
      float oo = __shfl_xor(d21[v], 8);
      h2[v] = lstm_h(d20[v], d21[v], oo);
    }
    lds_fence();
    if (m < 8) storeH(hb, 8, m, g, h2);
    lds_fence();
    a = loadA8(hb, m, g);

    // ---- layer 3 (8 -> 16)
    d_i = wmma16(a, sB[5 * 32 + lane], cfill(b3i));
    d_g = wmma16(a, sB[6 * 32 + lane], cfill(b3g));
    d_o = wmma16(a, sB[7 * 32 + lane], cfill(b3o));
    v8f h3 = combine3(d_i, d_g, d_o);
    storeH(hb, 16, m, g, h3);
    lds_fence();
    a = loadA16(hb, m, g);

    // ---- layer 4 (16 -> 32): two passes of {i,g,o} to cap live accumulators
    v8f e0 = wmma16(a, sB[8 * 32 + lane],  cfill(b4[0]));
    v8f e2 = wmma16(a, sB[10 * 32 + lane], cfill(b4[2]));
    v8f e4 = wmma16(a, sB[12 * 32 + lane], cfill(b4[4]));
    v8f h40 = combine3(e0, e2, e4);
    v8f e1 = wmma16(a, sB[9 * 32 + lane],  cfill(b4[1]));
    v8f e3 = wmma16(a, sB[11 * 32 + lane], cfill(b4[3]));
    v8f e5 = wmma16(a, sB[13 * 32 + lane], cfill(b4[5]));
    v8f h41 = combine3(e1, e3, e5);

    // ---- output head: out[r] = dot(h4[r], w_out) + b_out
    v8f s;
#pragma unroll
    for (int v = 0; v < 8; ++v) {
      float t = h40[v] * wA + h41[v] * wB;
      t += __shfl_xor(t, 1);
      t += __shfl_xor(t, 2);
      t += __shfl_xor(t, 4);
      t += __shfl_xor(t, 8);
      s[v] = t;  // row (v + 8g) sum, replicated in all 16 lanes of the half
    }
    float r = s[0];
#pragma unroll
    for (int v = 1; v < 8; ++v) r = (m == v) ? s[v] : r;
    if (m < 8) out[base + m + 8 * g] = r + bO;
  }
}

extern "C" void kernel_launch(void* const* d_in, const int* in_sizes, int n_in,
                              void* d_out, int out_size, void* d_ws,
                              size_t ws_size, hipStream_t stream) {
  (void)n_in; (void)out_size; (void)d_ws; (void)ws_size;
  const float* x    = (const float*)d_in[0];
  const float* wih0 = (const float*)d_in[1];
  const float* bih0 = (const float*)d_in[3];
  const float* bhh0 = (const float*)d_in[4];
  const float* wih1 = (const float*)d_in[5];
  const float* bih1 = (const float*)d_in[7];
  const float* bhh1 = (const float*)d_in[8];
  const float* wih2 = (const float*)d_in[9];
  const float* bih2 = (const float*)d_in[11];
  const float* bhh2 = (const float*)d_in[12];
  const float* wih3 = (const float*)d_in[13];
  const float* bih3 = (const float*)d_in[15];
  const float* bhh3 = (const float*)d_in[16];
  const float* wih4 = (const float*)d_in[17];
  const float* bih4 = (const float*)d_in[19];
  const float* bhh4 = (const float*)d_in[20];
  const float* wout = (const float*)d_in[21];
  const float* bout = (const float*)d_in[22];
  float* out = (float*)d_out;

  const int n = in_sizes[0];
  int nTiles = n / 16;
  int blocks = (nTiles + 7) / 8;   // 8 waves per block
  if (blocks > 2048) blocks = 2048;
  if (blocks < 1) blocks = 1;

  lstm_stack_kernel<<<blocks, TPB, 0, stream>>>(
      x, wih0, bih0, bhh0, wih1, bih1, bhh1, wih2, bih2, bhh2, wih3, bih3,
      bhh3, wih4, bih4, bhh4, wout, bout, out, nTiles);
}
